// BoostingNet_11811160064594
// MI455X (gfx1250) — matrix operand order repaired and checked by reading, not compiled, not run on hardware
//
#include <hip/hip_runtime.h>

// CDNA5 / gfx1250 LSTM forward: 4 layers, H=32, T=64, B=16384.
// Compute-bound (~60 GFLOP vs ~13 MB HBM). One v_wmma_f32_16x16x32_f16 per
// 16x16 gate tile (K=32 == HIDDEN). Elementwise uses hardware v_tanh_f32.
// Trick this round: stage h2 = 2*h (h2 = fma(tanh_o, tanh_c, tanh_c)) and
// pre-HALVE all f16 weights in LDS (and W_lin in the epilogue), removing the
// trailing 0.5 multiply from every state element. inpA copy removed
// (layer l input == hA[l-1]).

typedef __attribute__((ext_vector_type(16))) _Float16 v16h;
typedef __attribute__((ext_vector_type(8)))  _Float16 v8h;
typedef __attribute__((ext_vector_type(8)))  float    v8f;

#define NL 4
#define HID 32
#define GATES 128
#define TSTEPS 64
#define BATCH 16384

// ---- activations: gfx1250 V_TANH_F32 (TRANS, co-executes with XDL WMMA)
#if __has_builtin(__builtin_amdgcn_tanhf)
__device__ __forceinline__ float dtanh(float x) { return __builtin_amdgcn_tanhf(x); }
#elif __has_builtin(__builtin_amdgcn_tanh_f32)
__device__ __forceinline__ float dtanh(float x) { return __builtin_amdgcn_tanh_f32(x); }
#else
__device__ __forceinline__ float dtanh(float x) {
    float e = __builtin_amdgcn_exp2f(2.88539008177792681472f * x);  // exp(2x)
    return 1.0f - 2.0f * __builtin_amdgcn_rcpf(e + 1.0f);
}
#endif

// sigmoid(x + b) with pre-halved bias hb = 0.5*b:  fma + v_tanh + fma
__device__ __forceinline__ float sig_pb(float x, float hb) {
    return __builtin_fmaf(0.5f, dtanh(__builtin_fmaf(0.5f, x, hb)), 0.5f);
}
__device__ __forceinline__ float dsig(float x) {
    return __builtin_fmaf(0.5f, dtanh(0.5f * x), 0.5f);
}

__device__ __forceinline__ v8f wmma_f16(v16h a, v16h b, v8f c) {
    return __builtin_amdgcn_wmma_f32_16x16x32_f16(false, a, false, b,
                                                  (short)0, c, false, false);
}

// Load one pre-swizzled 32B fragment chunk: two ds_load_b128, imm offsets.
__device__ __forceinline__ v16h ldfrag(const _Float16* p) {
    v8h lo = *(const v8h*)(p);
    v8h hi = *(const v8h*)(p + 8);
    v16h r;
#pragma unroll
    for (int i = 0; i < 8; ++i) { r[i] = lo[i]; r[i + 8] = hi[i]; }
    return r;
}

__global__ __launch_bounds__(128)
void lstm4_wmma_kernel(const float* __restrict__ x,      // (B,3,T)
                       const float* __restrict__ Wih0,   // (128,2)
                       const float* __restrict__ Wih,    // (3,128,32)
                       const float* __restrict__ Whh,    // (4,128,32)
                       const float* __restrict__ bih,    // (4,128)
                       const float* __restrict__ bhh,    // (4,128)
                       const float* __restrict__ Wlin,   // (1,32)
                       const float* __restrict__ blin,   // (1,)
                       float* __restrict__ out)          // (B,)
{
    // Fragment-swizzled weights: [tile = l*8+gt][lane][16 halfs] -> each lane's
    // B-operand chunk is 32 contiguous bytes (2x ds_load_b128, imm offset).
    // NOTE: stored as 0.5*W because the staged hidden state is h2 = 2*h.
    __shared__ __align__(16) _Float16 sWhhF[NL * 8 * 32 * 16];        // 32 KB
    __shared__ __align__(16) _Float16 sWihF[(NL - 1) * 8 * 32 * 16];  // 24 KB
    __shared__ __align__(16) float    sBias[NL * GATES];              // 2 KB
    __shared__ __align__(16) float    sWih0[GATES * 2];               // 1 KB
    __shared__ __align__(16) _Float16 sStage[4 * 32 * 16];            // 4 KB (1 KB/wave)

    const int tid = threadIdx.x;

    // ---- one-time preload: swizzle f32 weights into (halved) f16 fragments --
    // frag element i of lane ln, tile (l,gt):  k = (ln>>4)*8 + (i&7) + (i>>3)*16, n = gt*16 + (ln&15)
    for (int d = tid; d < NL * 8 * 32 * 16; d += 128) {
        int i  = d & 15;
        int ln = (d >> 4) & 31;
        int gt = (d >> 9) & 7;
        int l  = d >> 12;
        int k  = ((ln >> 4) << 3) + (i & 7) + ((i >> 3) << 4);
        int n  = (gt << 4) + (ln & 15);
        sWhhF[d] = (_Float16)(0.5f * Whh[l * (GATES * HID) + n * HID + k]);
    }
    for (int d = tid; d < (NL - 1) * 8 * 32 * 16; d += 128) {
        int i  = d & 15;
        int ln = (d >> 4) & 31;
        int gt = (d >> 9) & 7;
        int l  = d >> 12;
        int k  = ((ln >> 4) << 3) + (i & 7) + ((i >> 3) << 4);
        int n  = (gt << 4) + (ln & 15);
        sWihF[d] = (_Float16)(0.5f * Wih[l * (GATES * HID) + n * HID + k]);
    }
    for (int i = tid; i < NL * GATES; i += 128) sBias[i] = bih[i] + bhh[i];
    for (int i = tid; i < GATES * 2; i += 128) sWih0[i] = Wih0[i];
    __syncthreads();

    const int lane = tid & 31;
    const int wave = tid >> 5;
    const int half = lane >> 4;        // 0: lanes 0-15, 1: lanes 16-31
    const int li   = lane & 15;
    const int b0   = blockIdx.x * 64 + wave * 16;

    // Per-lane fragment base pointers (tile offsets become immediates).
    const _Float16* wB = sWhhF + lane * 16;
    const _Float16* iB = sWihF + lane * 16;
    // A-staging: [wave][lane][16 halfs]; scatter-store base for D-layout h:
    // dest lane = m + 16*((j>>3)&1), slot = (j&7) + 8*(j>>4); with
    // m = r + 8*half, j = jt*16 + li this collapses to base + (r*16 + 8*jt).
    const int hb = (li >> 3) & 1;
    _Float16* stW = sStage + wave * 512 + ((half << 3) + (hb << 4)) * 16 + (li & 7);
    const _Float16* stR = sStage + wave * 512 + lane * 16;

    // Pre-halved sigmoid biases + raw g-gate bias, per lane (loop-invariant).
    float bI[NL][2], bF[NL][2], bG[NL][2], bO[NL][2];
#pragma unroll
    for (int l = 0; l < NL; ++l)
#pragma unroll
        for (int jt = 0; jt < 2; ++jt) {
            int j = jt * 16 + li;
            bI[l][jt] = 0.5f * sBias[l * GATES + 0  + j];
            bF[l][jt] = 0.5f * sBias[l * GATES + 32 + j];
            bG[l][jt] =        sBias[l * GATES + 64 + j];
            bO[l][jt] = 0.5f * sBias[l * GATES + 96 + j];
        }
    // Layer-0 input weights per lane (2-wide input -> VALU FMAs, not WMMA).
    float w0v[8], w1v[8];
#pragma unroll
    for (int gt = 0; gt < 8; ++gt) {
        w0v[gt] = sWih0[((gt << 4) + li) * 2 + 0];
        w1v[gt] = sWih0[((gt << 4) + li) * 2 + 1];
    }

    // Recurrent state. c (true scale, f32) in C/D layout: (m=r+8*half, j=jt*16+li)
    float cst[NL][16];
    v16h  hA[NL];                       // 2*h in A-operand f16 layout
#pragma unroll
    for (int l = 0; l < NL; ++l) {
#pragma unroll
        for (int i = 0; i < 16; ++i) cst[l][i] = 0.0f;
#pragma unroll
        for (int i = 0; i < 16; ++i) hA[l][i] = (_Float16)0.0f;
    }

    const float* xrow = x + (size_t)(b0 + 8 * half) * (3 * TSTEPS);

#pragma unroll 1
    for (int t = 0; t < TSTEPS; ++t) {
        float x0[8], x1[8];
#pragma unroll
        for (int r = 0; r < 8; ++r) {
            x0[r] = xrow[r * (3 * TSTEPS) + t];
            x1[r] = xrow[r * (3 * TSTEPS) + TSTEPS + t];
        }

#pragma unroll
        for (int l = 0; l < NL; ++l) {
            // two phases: gate tiles {jt,2+jt,4+jt,6+jt} = i/f/g/o column-half
#pragma unroll
            for (int jt = 0; jt < 2; ++jt) {
                v8f a[4];
#pragma unroll
                for (int q = 0; q < 4; ++q) {
                    const int gt = 2 * q + jt;
                    v8f acc = {};                      // WMMA C = inline 0
                    if (l == 0) {
#pragma unroll
                        for (int r = 0; r < 8; ++r)
                            acc[r] = __builtin_fmaf(x1[r], w1v[gt], x0[r] * w0v[gt]);
                    } else {
                        // input = h2 of layer l-1 (current t), weights pre-halved
                        acc = wmma_f16(hA[l - 1], ldfrag(iB + ((l - 1) * 8 + gt) * 512), acc);
                    }
                    acc = wmma_f16(hA[l], ldfrag(wB + (l * 8 + gt) * 512), acc);
                    a[q] = acc;
                }
                // elementwise cell, bias folded into activation FMAs
#pragma unroll
                for (int r = 0; r < 8; ++r) {
                    float iv = sig_pb(a[0][r], bI[l][jt]);
                    float fv = sig_pb(a[1][r], bF[l][jt]);
                    float gv = dtanh(a[2][r] + bG[l][jt]);
                    float to = dtanh(__builtin_fmaf(0.5f, a[3][r], bO[l][jt]));
                    int   ci = jt * 8 + r;
                    float cn = __builtin_fmaf(fv, cst[l][ci], iv * gv);
                    cst[l][ci] = cn;
                    float tc = dtanh(cn);
                    // h2 = (tanh_o + 1) * tanh_c = 2 * sigmoid(o)*tanh(c)
                    float h2 = __builtin_fmaf(to, tc, tc);
                    stW[r * 16 + 8 * jt] = (_Float16)h2;   // imm-offset b16 store
                }
            }
            // per-wave private staging: DS ops in-order within wave, compiler
            // inserts s_wait_dscnt; no barrier in the hot loop.
            hA[l] = ldfrag(stR);
        }
    }

    // ---- epilogue: out[b] = sigmoid(h3 . Wlin + blin), staged h3 is 2*h ----
    // sStage holds layer-3 h2 (f16, fragment layout):
    // lane m row: chunk m = {j 0-7, j 16-23}, chunk 16+m = {j 8-15, j 24-31}.
    if (lane < 16) {
        const _Float16* p0 = sStage + wave * 512 + lane * 16;
        const _Float16* p1 = sStage + wave * 512 + (16 + lane) * 16;
        float dot = blin[0];
#pragma unroll
        for (int q = 0; q < 8; ++q) {
            dot = __builtin_fmaf((float)p0[q],     0.5f * Wlin[q],      dot);
            dot = __builtin_fmaf((float)p1[q],     0.5f * Wlin[8 + q],  dot);
            dot = __builtin_fmaf((float)p0[8 + q], 0.5f * Wlin[16 + q], dot);
            dot = __builtin_fmaf((float)p1[8 + q], 0.5f * Wlin[24 + q], dot);
        }
        out[b0 + lane] = dsig(dot);
    }
}

extern "C" void kernel_launch(void* const* d_in, const int* in_sizes, int n_in,
                              void* d_out, int out_size, void* d_ws, size_t ws_size,
                              hipStream_t stream) {
    (void)in_sizes; (void)n_in; (void)d_ws; (void)ws_size; (void)out_size;
    const float* x    = (const float*)d_in[0];
    const float* Wih0 = (const float*)d_in[1];
    const float* Wih  = (const float*)d_in[2];
    const float* Whh  = (const float*)d_in[3];
    const float* bih  = (const float*)d_in[4];
    const float* bhh  = (const float*)d_in[5];
    const float* Wlin = (const float*)d_in[6];
    const float* blin = (const float*)d_in[7];
    float* out = (float*)d_out;

    dim3 grid(BATCH / 64);   // 256 blocks; 4 waves/block, 16 batch rows/wave
    dim3 block(128);
    lstm4_wmma_kernel<<<grid, block, 0, stream>>>(x, Wih0, Wih, Whh, bih, bhh,
                                                  Wlin, blin, out);
}
